// LightweightMultiScaleCrossAttn_28398323761488
// MI455X (gfx1250) — matrix operand order repaired and checked by reading, not compiled, not run on hardware
//
#include <hip/hip_runtime.h>
#include <hip/hip_bf16.h>
#include <math.h>

// ---------------------------------------------------------------------------
// Types for CDNA5 WMMA
// ---------------------------------------------------------------------------
typedef __bf16 bf16_t;
typedef __attribute__((ext_vector_type(16))) __bf16 v16bf;
typedef __attribute__((ext_vector_type(8)))  __bf16 v8bf;
typedef __attribute__((ext_vector_type(8)))  float  v8f;
typedef __attribute__((ext_vector_type(4)))  float  v4f;

#define LDIM   2048
#define LNEPS  1e-5f

// round-to-nearest-even fp32 -> bf16 bits (no reliance on __bf16 conversions)
__device__ __forceinline__ unsigned short f2bfbits(float f) {
    unsigned u = __builtin_bit_cast(unsigned, f);
    unsigned r = 0x7FFFu + ((u >> 16) & 1u);
    return (unsigned short)((u + r) >> 16);
}
__device__ __forceinline__ bf16_t bits2bf(unsigned short s) {
    return __builtin_bit_cast(bf16_t, s);
}
__device__ __forceinline__ bf16_t f2bf(float f) { return bits2bf(f2bfbits(f)); }

__device__ __forceinline__ v8f wmma_bf16(v16bf a, v16bf b, v8f c) {
    // D = A(16x32 bf16) * B(32x16 bf16) + C(16x16 f32)
    return __builtin_amdgcn_wmma_f32_16x16x32_bf16(
        false, a, false, b, (short)0, c, false, false);
}

// ---------------------------------------------------------------------------
// Fragment loaders (ISA 7.12.2 layouts, wave32).
// A frag (16x32): lane[3:0]=M, lane[4]=kh. element t: k = (t<8 ? kh*8+t
//                 : 16+kh*8+(t-8))  -> two contiguous 8-element chunks.
// B frag (32x16): lane[3:0]=N, lane[4]=kh. element t: same k as A frag
//                 -> two contiguous 8-element chunks.
// ---------------------------------------------------------------------------
__device__ __forceinline__ v16bf combine88(v8bf lo, v8bf hi) {
    v16bf a;
#pragma unroll
    for (int j = 0; j < 8; ++j) { a[j] = lo[j]; a[8 + j] = hi[j]; }
    return a;
}
// A fragment from global bf16 row-major [rows][lda]
__device__ __forceinline__ v16bf load_a_bf16g(const unsigned short* A, int lda, int r0, int c0) {
    int lane = threadIdx.x & 31, m = lane & 15, kh = lane >> 4;
    const unsigned short* p = A + (size_t)(r0 + m) * lda + c0;
    v8bf lo = *(const v8bf*)(p + kh * 8);
    v8bf hi = *(const v8bf*)(p + 16 + kh * 8);
    return combine88(lo, hi);
}
// A fragment from LDS bf16 (row stride ld, 16B-aligned base+cols)
__device__ __forceinline__ v16bf load_a_lds_bf16(const unsigned short* s, int ld) {
    int lane = threadIdx.x & 31, m = lane & 15, kh = lane >> 4;
    const unsigned short* p = s + m * ld;
    v8bf lo = *(const v8bf*)(p + kh * 8);
    v8bf hi = *(const v8bf*)(p + 16 + kh * 8);
    return combine88(lo, hi);
}
// A fragment from LDS f32 16x32 tile (convert at load)
__device__ __forceinline__ v16bf load_a_lds_f32(const float* s) {
    int lane = threadIdx.x & 31, m = lane & 15, kh = lane >> 4;
    const float* p = s + m * 32;
    v4f x0 = *(const v4f*)(p + kh * 8);
    v4f x1 = *(const v4f*)(p + kh * 8 + 4);
    v4f x2 = *(const v4f*)(p + 16 + kh * 8);
    v4f x3 = *(const v4f*)(p + 16 + kh * 8 + 4);
    v16bf a;
#pragma unroll
    for (int j = 0; j < 4; ++j) {
        a[j]      = f2bf(x0[j]);
        a[4 + j]  = f2bf(x1[j]);
        a[8 + j]  = f2bf(x2[j]);
        a[12 + j] = f2bf(x3[j]);
    }
    return a;
}
// B fragment: element (k,n) = B[(n_r0+n)*rowstride + k0 + k] (k contiguous)
__device__ __forceinline__ v16bf load_b_bf16g(const unsigned short* B, size_t rowstride,
                                              int n_r0, int k0) {
    int lane = threadIdx.x & 31, n = lane & 15, kh = lane >> 4;
    const unsigned short* p = B + (size_t)(n_r0 + n) * rowstride + k0;
    v8bf lo = *(const v8bf*)(p + kh * 8);
    v8bf hi = *(const v8bf*)(p + 16 + kh * 8);
    return combine88(lo, hi);
}
// C/D (16x16 f32): VGPR i -> row = i + 8*lane[4], col = lane[3:0]
__device__ __forceinline__ void store_c_lds(float* buf, int ld, int col0, v8f c, float scale) {
    int lane = threadIdx.x & 31;
    int n = lane & 15, rb = (lane >> 4) * 8;
#pragma unroll
    for (int i = 0; i < 8; ++i) buf[(rb + i) * ld + col0 + n] = c[i] * scale;
}

// output helper: out_mode 0 = f32 row-major, 1 = bf16 row-major, 2 = bf16 transposed
__device__ __forceinline__ void write_out(float* out, unsigned short* outb, int out_mode,
                                          size_t row, size_t col, int ldo, float v) {
    if (out_mode == 0)      out[row * ldo + col] = v;
    else if (out_mode == 1) outb[row * ldo + col] = f2bfbits(v);
    else                    outb[col * (size_t)ldo + row] = f2bfbits(v);
}

// ---------------------------------------------------------------------------
// Weight transpose+convert: Wt[n][k] = bf16(W[k][n])
// ---------------------------------------------------------------------------
__global__ void convT_kernel(const float* __restrict__ W, unsigned short* __restrict__ Wt,
                             int K, int N) {
    int idx = blockIdx.x * 256 + threadIdx.x;
    if (idx >= K * N) return;
    int k = idx / N, n = idx - k * N;
    Wt[(size_t)n * K + k] = f2bfbits(W[idx]);
}

// ---------------------------------------------------------------------------
// Generic GEMM + epilogue:  out = epilogue(A[16-row blk x K] @ Wt^T + bias)
//   Wt is bf16 [N][K] (pre-transposed).  A is f32, async-DMA staged to LDS.
//   NTW (template) = N/128 n-tiles per wave -> straight-line WMMA inner loop.
//   ln_mode: 0 none | 1 full-row LN (gamma/beta) | 2 per-32 LN no-affine
//   has_res: x = alpha*prev + beta*x before LN    leaky: leaky_relu(0.01)
// block = 256 threads (8 waves), grid.x = M/16. N in {256,512}, K % 32 == 0.
// ---------------------------------------------------------------------------
template <int NTW>
__global__ __launch_bounds__(256) void gemm_ep(
    const float* __restrict__ A, const unsigned short* __restrict__ Wt,
    const float* __restrict__ bias, const float* __restrict__ prev,
    const float* alpha_p, const float* beta_p,
    const float* __restrict__ gamma, const float* __restrict__ betaLN,
    float* __restrict__ out, unsigned short* __restrict__ outb,
    int K, int lda, int ldo, int out_row0, int out_col0,
    int ln_mode, int leaky, int has_res, int out_mode)
{
    constexpr int N = NTW * 128;

    __shared__ float Asf[16 * 32];
    __shared__ float rowbuf[16 * N];
    __shared__ float red1[256], red2[256];
    __shared__ float stats[32];

    const int tid  = threadIdx.x;
    const int wv   = tid >> 5;
    const int row0 = blockIdx.x * 16;

    v8f acc[NTW];
#pragma unroll
    for (int t = 0; t < NTW; ++t) acc[t] = (v8f){};

    for (int k0 = 0; k0 < K; k0 += 32) {
        // prefetch next A tile (global_prefetch_b8 on gfx1250)
        if (k0 + 32 < K && tid < 16)
            __builtin_prefetch(A + (size_t)(row0 + tid) * lda + k0 + 32, 0, 1);

        // async DMA the 16x32 f32 A tile into LDS (wave 0 issues; ASYNCcnt)
        if (wv == 0) {
            int lane = tid & 31;
            int rr0 = lane >> 3;           // 0..3
            int cc  = (lane & 7) * 4;      // 0,4,...,28
            unsigned ldsbase = (unsigned)(uintptr_t)Asf;
#pragma unroll
            for (int rr = 0; rr < 4; ++rr) {
                int row = rr * 4 + rr0;
                unsigned ldsa = ldsbase + (unsigned)((row * 32 + cc) * 4);
                const float* gp = A + (size_t)(row0 + row) * lda + k0 + cc;
                asm volatile("global_load_async_to_lds_b128 %0, %1, off"
                             :: "v"(ldsa), "v"(gp) : "memory");
            }
            asm volatile("s_wait_asynccnt 0" ::: "memory");
        }
        __syncthreads();

        v16bf a = load_a_lds_f32(Asf);
#pragma unroll
        for (int t = 0; t < NTW; ++t) {
            int n0 = (wv + t * 8) * 16;
            v16bf b = load_b_bf16g(Wt, K, n0, k0);
            acc[t] = wmma_bf16(a, b, acc[t]);
        }
        __syncthreads();
    }

    // dump accumulators + bias into rowbuf
    {
        int lane = tid & 31, n = lane & 15, rb = (lane >> 4) * 8;
#pragma unroll
        for (int t = 0; t < NTW; ++t) {
            int n0 = (wv + t * 8) * 16;
#pragma unroll
            for (int i = 0; i < 8; ++i)
                rowbuf[(rb + i) * N + n0 + n] = acc[t][i] + bias[n0 + n];
        }
    }
    __syncthreads();

    if (has_res) {
        float al = *alpha_p, be = *beta_p;
        for (int e = tid; e < 16 * N; e += 256) {
            int r = e / N, c = e % N;
            rowbuf[r * N + c] = al * prev[(size_t)(row0 + r) * 256 + c] + be * rowbuf[r * N + c];
        }
        __syncthreads();
    }

    if (ln_mode == 1) {
        int r = tid >> 4, seg = tid & 15;
        constexpr int sw = N / 16;
        float s = 0.f, s2 = 0.f;
        for (int c = seg * sw; c < seg * sw + sw; ++c) {
            float v = rowbuf[r * N + c];
            s += v; s2 += v * v;
        }
        red1[r * 16 + seg] = s; red2[r * 16 + seg] = s2;
        __syncthreads();
        if (seg == 0) {
            float S = 0.f, S2 = 0.f;
            for (int k = 0; k < 16; ++k) { S += red1[r * 16 + k]; S2 += red2[r * 16 + k]; }
            float mean = S / N, var = S2 / N - mean * mean;
            stats[r * 2] = mean; stats[r * 2 + 1] = rsqrtf(var + LNEPS);
        }
        __syncthreads();
        for (int e = tid; e < 16 * N; e += 256) {
            int rr = e / N, c = e % N;
            float v = (rowbuf[rr * N + c] - stats[rr * 2]) * stats[rr * 2 + 1];
            v = v * gamma[c] + betaLN[c];
            if (leaky) v = (v >= 0.f) ? v : 0.01f * v;
            write_out(out, outb, out_mode, (size_t)(out_row0 + row0 + rr),
                      (size_t)(out_col0 + c), ldo, v);
        }
    } else if (ln_mode == 2) {
        constexpr int ngr = N / 32;
        if (tid < 16 * ngr) {
            int r = tid / ngr, g = tid % ngr;
            float s = 0.f, s2 = 0.f;
            for (int c = g * 32; c < g * 32 + 32; ++c) {
                float v = rowbuf[r * N + c];
                s += v; s2 += v * v;
            }
            float mean = s / 32.f, var = s2 / 32.f - mean * mean;
            float rs = rsqrtf(var + LNEPS);
            for (int c = g * 32; c < g * 32 + 32; ++c)
                write_out(out, outb, out_mode, (size_t)(out_row0 + row0 + r),
                          (size_t)(out_col0 + c), ldo, (rowbuf[r * N + c] - mean) * rs);
        }
    } else {
        for (int e = tid; e < 16 * N; e += 256) {
            int r = e / N, c = e % N;
            float v = rowbuf[r * N + c];
            if (leaky) v = (v >= 0.f) ? v : 0.01f * v;
            write_out(out, outb, out_mode, (size_t)(out_row0 + row0 + r),
                      (size_t)(out_col0 + c), ldo, v);
        }
    }
}

// ---------------------------------------------------------------------------
// kNN: one thread per query, serial top-16 by squared distance; emit bitmaps
// (64 u32 per row per scale; top-8 is a prefix of the ascending top-16).
// ---------------------------------------------------------------------------
__global__ void knn_kernel(const float* __restrict__ pos,
                           unsigned* __restrict__ bit8,
                           unsigned* __restrict__ bit16)
{
    int i = blockIdx.x * blockDim.x + threadIdx.x;
    if (i >= LDIM) return;
    float px = pos[i * 3], py = pos[i * 3 + 1], pz = pos[i * 3 + 2];
    float bd[16]; int bi[16];
#pragma unroll
    for (int t = 0; t < 16; ++t) { bd[t] = 3.4e38f; bi[t] = 0; }
    for (int j = 0; j < LDIM; ++j) {
        float dx = px - pos[j * 3], dy = py - pos[j * 3 + 1], dz = pz - pos[j * 3 + 2];
        float d2 = dx * dx + dy * dy + dz * dz;
        if (d2 < bd[15]) {
            int p = 15;
            while (p > 0 && bd[p - 1] > d2) { bd[p] = bd[p - 1]; bi[p] = bi[p - 1]; --p; }
            bd[p] = d2; bi[p] = j;
        }
    }
    unsigned w16[64], w8[64];
    for (int w = 0; w < 64; ++w) { w16[w] = 0u; w8[w] = 0u; }
    for (int t = 0; t < 16; ++t) {
        int j = bi[t];
        w16[j >> 5] |= (1u << (j & 31));
        if (t < 8) w8[j >> 5] |= (1u << (j & 31));
    }
    for (int w = 0; w < 64; ++w) {
        bit16[(size_t)i * 64 + w] = w16[w];
        bit8[(size_t)i * 64 + w]  = w8[w];
    }
}

// ---------------------------------------------------------------------------
// Fused dual-scale flash attention (bf16 operands, WMMA, online softmax with
// two (m,l) stat pairs: knn8 / knn16 masks on the sparse half).
// grid = (L/16, H, 2 families). block = 256 (8 waves).
// ---------------------------------------------------------------------------
#define ATTN_SMEM (16*256*4 + 2*16*256*2 + 2*16*64*4 + 6*16*4 + 8*16*64*4)

__global__ __launch_bounds__(256) void attn_kernel(
    const unsigned short* __restrict__ Qbg, const unsigned short* __restrict__ Qbs,
    const unsigned short* __restrict__ Kb,  const unsigned short* __restrict__ Vt,
    const unsigned* __restrict__ bit8, const unsigned* __restrict__ bit16,
    const float* __restrict__ sw_geo, const float* __restrict__ sw_sem,
    float* __restrict__ att)
{
    extern __shared__ char smem[];
    float*          sc    = (float*)smem;                      // [16][256]
    unsigned short* p8    = (unsigned short*)(sc + 16 * 256);  // [16][256] bf16
    unsigned short* p16   = p8 + 16 * 256;                     // [16][256] bf16
    unsigned*       bmp8  = (unsigned*)(p16 + 16 * 256);       // [16][64]
    unsigned*       bmp16 = bmp8 + 16 * 64;                    // [16][64]
    float*          m8s   = (float*)(bmp16 + 16 * 64);
    float*          l8s   = m8s + 16;
    float*          m16s  = l8s + 16;
    float*          l16s  = m16s + 16;
    float*          e8s   = l16s + 16;
    float*          e16s  = e8s + 16;
    float*          odump = e16s + 16;                         // [8][16][64]

    const int tid = threadIdx.x, wv = tid >> 5;
    const int q0  = blockIdx.x * 16;
    const int h   = blockIdx.y;
    const int fam = blockIdx.z;
    const float inv_sqrt = 0.17677669529663687f; // 1/sqrt(32)

    for (int e = tid; e < 16 * 64; e += 256) {
        int r = e >> 6, w = e & 63;
        bmp16[e] = bit16[(size_t)(q0 + r) * 64 + w];
        bmp8[e]  = bit8[(size_t)(q0 + r) * 64 + w];
    }
    if (tid < 16) {
        m8s[tid] = -1e30f; l8s[tid] = 0.f;
        m16s[tid] = -1e30f; l16s[tid] = 0.f;
    }
    const unsigned short* Qb = fam ? Qbs : Qbg;
    v16bf a_q = load_a_bf16g(Qb, 256, q0, h * 32);
    v8f o8lo = {}, o8hi = {}, o16lo = {}, o16hi = {};
    __syncthreads();

    for (int it = 0; it < 16; ++it) {
        int kbase = it * 256;
        // prefetch next chunk of keys
        if (it + 1 < 16 && tid < 32)
            __builtin_prefetch(Kb + (size_t)(kbase + 256 + tid * 8) * 256, 0, 1);

        // --- scores: each wave does 2 WMMAs (32 keys) -----------------------
#pragma unroll
        for (int half = 0; half < 2; ++half) {
            int key0 = kbase + wv * 32 + half * 16;
            v16bf b = load_b_bf16g(Kb, 256, key0, h * 32);
            v8f s = {};
            s = wmma_bf16(a_q, b, s);
            store_c_lds(sc, 256, wv * 32 + half * 16, s, inv_sqrt);
        }
        __syncthreads();

        // --- dual-scale online softmax stats (one thread per query row) ----
        if (tid < 16) {
            int r = tid;
            float cm8 = -1e30f, cm16 = -1e30f;
            for (int c = 0; c < 256; ++c) {
                int kg = kbase + c;
                bool inMasked = fam == 0 ? (kg < LDIM) : (kg >= LDIM);
                bool a16 = true, a8 = true;
                if (inMasked) {
                    int mi = (fam == 0) ? kg : kg - LDIM;
                    a16 = (bmp16[r * 64 + (mi >> 5)] >> (mi & 31)) & 1u;
                    a8  = (bmp8 [r * 64 + (mi >> 5)] >> (mi & 31)) & 1u;
                }
                float v = sc[r * 256 + c];
                if (a16) cm16 = fmaxf(cm16, v);
                if (a8)  cm8  = fmaxf(cm8, v);
            }
            float nm16 = fmaxf(m16s[r], cm16);
            float nm8  = fmaxf(m8s[r],  cm8);
            float f16v = __expf(m16s[r] - nm16);
            float f8v  = __expf(m8s[r]  - nm8);
            float s16 = 0.f, s8 = 0.f;
            for (int c = 0; c < 256; ++c) {
                int kg = kbase + c;
                bool inMasked = fam == 0 ? (kg < LDIM) : (kg >= LDIM);
                bool a16 = true, a8 = true;
                if (inMasked) {
                    int mi = (fam == 0) ? kg : kg - LDIM;
                    a16 = (bmp16[r * 64 + (mi >> 5)] >> (mi & 31)) & 1u;
                    a8  = (bmp8 [r * 64 + (mi >> 5)] >> (mi & 31)) & 1u;
                }
                float v = sc[r * 256 + c];
                float pv16 = a16 ? __expf(v - nm16) : 0.f;
                float pv8  = a8  ? __expf(v - nm8)  : 0.f;
                s16 += pv16; s8 += pv8;
                p16[r * 256 + c] = f2bfbits(pv16);
                p8 [r * 256 + c] = f2bfbits(pv8);
            }
            l16s[r] = l16s[r] * f16v + s16; m16s[r] = nm16; e16s[r] = f16v;
            l8s[r]  = l8s[r]  * f8v  + s8;  m8s[r]  = nm8;  e8s[r]  = f8v;
        }
        __syncthreads();

        // --- rescale running O and accumulate P*V (each wave: 32 keys) ------
        {
            int lane = tid & 31, rb = (lane >> 4) * 8;
#pragma unroll
            for (int i = 0; i < 8; ++i) {
                float f8v = e8s[rb + i], f16v = e16s[rb + i];
                o8lo[i] *= f8v;  o8hi[i] *= f8v;
                o16lo[i] *= f16v; o16hi[i] *= f16v;
            }
        }
        {
            int klocal = wv * 32;
            v16bf pa8  = load_a_lds_bf16(p8  + klocal, 256);
            v16bf pa16 = load_a_lds_bf16(p16 + klocal, 256);
            // B = V: element (k=key, n=d) = Vt[(d0+n)*4096 + key0+k] (contig keys)
            v16bf blo = load_b_bf16g(Vt, 4096, h * 32,      kbase + klocal);
            v16bf bhi = load_b_bf16g(Vt, 4096, h * 32 + 16, kbase + klocal);
            o8lo  = wmma_bf16(pa8,  blo, o8lo);
            o8hi  = wmma_bf16(pa8,  bhi, o8hi);
            o16lo = wmma_bf16(pa16, blo, o16lo);
            o16hi = wmma_bf16(pa16, bhi, o16hi);
        }
        __syncthreads();
    }

    // --- cross-wave reduce + scale-weight combine ---------------------------
    {
        float* od = odump + wv * 16 * 64;
        int lane = tid & 31, n = lane & 15, rb = (lane >> 4) * 8;
#pragma unroll
        for (int i = 0; i < 8; ++i) {
            int r = rb + i;
            od[r * 64 + n]      = o8lo[i];
            od[r * 64 + 16 + n] = o8hi[i];
            od[r * 64 + 32 + n] = o16lo[i];
            od[r * 64 + 48 + n] = o16hi[i];
        }
    }
    __syncthreads();
    {
        const float* sw = fam ? sw_sem : sw_geo;
        float s0 = sw[0], s1 = sw[1];
        float mx = fmaxf(s0, s1);
        float e0 = __expf(s0 - mx), e1 = __expf(s1 - mx);
        float w0 = e0 / (e0 + e1), w1 = e1 / (e0 + e1);
        for (int e = tid; e < 512; e += 256) {
            int r = e >> 5, d = e & 31;
            float a8 = 0.f, a16 = 0.f;
            for (int w = 0; w < 8; ++w) {
                a8  += odump[w * 1024 + r * 64 + d];
                a16 += odump[w * 1024 + r * 64 + 32 + d];
            }
            float val = w0 * a8 / l8s[r] + w1 * a16 / l16s[r];
            att[((size_t)fam * LDIM + q0 + r) * 256 + h * 32 + d] = val;
        }
    }
}

// ---------------------------------------------------------------------------
// Host-side orchestration
// ---------------------------------------------------------------------------
extern "C" void kernel_launch(void* const* d_in, const int* in_sizes, int n_in,
                              void* d_out, int out_size, void* d_ws, size_t ws_size,
                              hipStream_t stream) {
    (void)in_sizes; (void)n_in; (void)out_size; (void)ws_size;

    const float* geo_feat = (const float*)d_in[0];
    const float* sem_feat = (const float*)d_in[1];
    const float* rsa_feat = (const float*)d_in[2];
    const float* pos      = (const float*)d_in[3];
    const float* Wg  = (const float*)d_in[4];  const float* bg  = (const float*)d_in[5];
    const float* gg  = (const float*)d_in[6];  const float* gb  = (const float*)d_in[7];
    const float* Ws  = (const float*)d_in[8];  const float* bs  = (const float*)d_in[9];
    const float* sg  = (const float*)d_in[10]; const float* sb  = (const float*)d_in[11];
    const float* Wr  = (const float*)d_in[12]; const float* br  = (const float*)d_in[13];
    const float* rg  = (const float*)d_in[14]; const float* rb  = (const float*)d_in[15];
    const float* Wqg = (const float*)d_in[16]; const float* bqg = (const float*)d_in[17];
    const float* Wqs = (const float*)d_in[18]; const float* bqs = (const float*)d_in[19];
    const float* Wk  = (const float*)d_in[20]; const float* bk  = (const float*)d_in[21];
    const float* Wv  = (const float*)d_in[22]; const float* bv  = (const float*)d_in[23];
    const float* Wo  = (const float*)d_in[24]; const float* bo  = (const float*)d_in[25];
    const float* sw_geo = (const float*)d_in[26]; const float* sw_sem = (const float*)d_in[27];
    const float* alpha_geo = (const float*)d_in[28]; const float* beta_geo = (const float*)d_in[29];
    const float* alpha_sem = (const float*)d_in[30]; const float* beta_sem = (const float*)d_in[31];
    const float* Wrt = (const float*)d_in[32]; const float* brt = (const float*)d_in[33];
    const float* rtg = (const float*)d_in[34]; const float* rtb = (const float*)d_in[35];
    const float* Wf1 = (const float*)d_in[36]; const float* bf1 = (const float*)d_in[37];
    const float* f1g = (const float*)d_in[38]; const float* f1b = (const float*)d_in[39];
    const float* Wf2 = (const float*)d_in[40]; const float* bf2 = (const float*)d_in[41];
    const float* f2g = (const float*)d_in[42]; const float* f2b = (const float*)d_in[43];
    const float* ln1g = (const float*)d_in[44]; const float* ln1b = (const float*)d_in[45];
    const float* ln2g = (const float*)d_in[46]; const float* ln2b = (const float*)d_in[47];

    // ---- workspace carve-up ------------------------------------------------
    float* ws = (float*)d_ws;
    float* geo_p    = ws;                              // 2048*256 f32
    float* sem_p    = geo_p + 2048 * 256;
    float* rsa_p    = sem_p + 2048 * 256;
    float* attbuf   = rsa_p + 2048 * 256;              // 2*2048*256 f32
    float* combined = attbuf + 2 * 2048 * 256;         // 2048*768 f32
    float* h1       = combined + 2048 * 768;           // 2048*512 f32
    unsigned short* Kb  = (unsigned short*)(h1 + 2048 * 512);  // [4096][256] bf16
    unsigned short* Vt  = Kb + 4096 * 256;                     // [256][4096] bf16 (transposed)
    unsigned short* Qbg = Vt + 4096 * 256;                     // [2048][256] bf16
    unsigned short* Qbs = Qbg + 2048 * 256;
    unsigned short* Wgt  = Qbs + 2048 * 256;           // bf16 transposed weights [N][K]
    unsigned short* Wst  = Wgt  + 1536 * 256;
    unsigned short* Wrt_t= Wst  + 512 * 256;
    unsigned short* Wqgt = Wrt_t+ 64 * 256;
    unsigned short* Wqst = Wqgt + 256 * 256;
    unsigned short* Wkt  = Wqst + 256 * 256;
    unsigned short* Wvt  = Wkt  + 256 * 256;
    unsigned short* Wot  = Wvt  + 256 * 256;
    unsigned short* Wrtt = Wot  + 256 * 256;
    unsigned short* Wf1t = Wrtt + 256 * 256;
    unsigned short* Wf2t = Wf1t + 768 * 512;
    unsigned* bit8  = (unsigned*)(Wf2t + 512 * 256);   // 2048*64 u32
    unsigned* bit16 = bit8 + 2048 * 64;

    dim3 blk(256);
    dim3 g128(2048 / 16);
    auto cgrid = [](int n) { return dim3((unsigned)((n + 255) / 256)); };

    // 0) one-time weight transpose+convert to bf16 [N][K]
    convT_kernel<<<cgrid(1536 * 256), blk, 0, stream>>>(Wg,  Wgt,  1536, 256);
    convT_kernel<<<cgrid(512 * 256),  blk, 0, stream>>>(Ws,  Wst,  512,  256);
    convT_kernel<<<cgrid(64 * 256),   blk, 0, stream>>>(Wr,  Wrt_t,64,   256);
    convT_kernel<<<cgrid(256 * 256),  blk, 0, stream>>>(Wqg, Wqgt, 256,  256);
    convT_kernel<<<cgrid(256 * 256),  blk, 0, stream>>>(Wqs, Wqst, 256,  256);
    convT_kernel<<<cgrid(256 * 256),  blk, 0, stream>>>(Wk,  Wkt,  256,  256);
    convT_kernel<<<cgrid(256 * 256),  blk, 0, stream>>>(Wv,  Wvt,  256,  256);
    convT_kernel<<<cgrid(256 * 256),  blk, 0, stream>>>(Wo,  Wot,  256,  256);
    convT_kernel<<<cgrid(256 * 256),  blk, 0, stream>>>(Wrt, Wrtt, 256,  256);
    convT_kernel<<<cgrid(768 * 512),  blk, 0, stream>>>(Wf1, Wf1t, 768,  512);
    convT_kernel<<<cgrid(512 * 256),  blk, 0, stream>>>(Wf2, Wf2t, 512,  256);

    // 1) input projections + row-LN(affine)  (f32 out)
    gemm_ep<2><<<g128, blk, 0, stream>>>(geo_feat, Wgt, bg, nullptr, nullptr, nullptr, gg, gb,
                                         geo_p, nullptr, 1536, 1536, 256, 0, 0, 1, 0, 0, 0);
    gemm_ep<2><<<g128, blk, 0, stream>>>(sem_feat, Wst, bs, nullptr, nullptr, nullptr, sg, sb,
                                         sem_p, nullptr, 512, 512, 256, 0, 0, 1, 0, 0, 0);
    gemm_ep<2><<<g128, blk, 0, stream>>>(rsa_feat, Wrt_t, br, nullptr, nullptr, nullptr, rg, rb,
                                         rsa_p, nullptr, 64, 64, 256, 0, 0, 1, 0, 0, 0);

    // 2) K (bf16 row-major) / V (bf16 transposed) with per-head LN(no affine)
    gemm_ep<2><<<g128, blk, 0, stream>>>(geo_p, Wkt, bk, nullptr, nullptr, nullptr, nullptr, nullptr,
                                         nullptr, Kb, 256, 256, 256, 0, 0, 2, 0, 0, 1);
    gemm_ep<2><<<g128, blk, 0, stream>>>(sem_p, Wkt, bk, nullptr, nullptr, nullptr, nullptr, nullptr,
                                         nullptr, Kb, 256, 256, 256, 2048, 0, 2, 0, 0, 1);
    gemm_ep<2><<<g128, blk, 0, stream>>>(geo_p, Wvt, bv, nullptr, nullptr, nullptr, nullptr, nullptr,
                                         nullptr, Vt, 256, 256, 4096, 0, 0, 2, 0, 0, 2);
    gemm_ep<2><<<g128, blk, 0, stream>>>(sem_p, Wvt, bv, nullptr, nullptr, nullptr, nullptr, nullptr,
                                         nullptr, Vt, 256, 256, 4096, 2048, 0, 2, 0, 0, 2);

    // 3) queries (bias only, bf16 row-major out)
    gemm_ep<2><<<g128, blk, 0, stream>>>(geo_p, Wqgt, bqg, nullptr, nullptr, nullptr, nullptr, nullptr,
                                         nullptr, Qbg, 256, 256, 256, 0, 0, 0, 0, 0, 1);
    gemm_ep<2><<<g128, blk, 0, stream>>>(sem_p, Wqst, bqs, nullptr, nullptr, nullptr, nullptr, nullptr,
                                         nullptr, Qbs, 256, 256, 256, 0, 0, 0, 0, 0, 1);

    // 4) kNN bitmaps (k=8 prefix of k=16)
    knn_kernel<<<dim3(8), blk, 0, stream>>>(pos, bit8, bit16);

    // 5) fused dual-scale flash attention (both families, all heads)
    attn_kernel<<<dim3(128, 8, 2), blk, ATTN_SMEM, stream>>>(
        Qbg, Qbs, Kb, Vt, bit8, bit16, sw_geo, sw_sem, attbuf);

    // 6) Wo projection + residual + LN  -> combined[:, 0:256] / [:, 256:512]
    gemm_ep<2><<<g128, blk, 0, stream>>>(attbuf, Wot, bo, geo_p, alpha_geo, beta_geo, ln1g, ln1b,
                                         combined, nullptr, 256, 256, 768, 0, 0, 1, 0, 1, 0);
    gemm_ep<2><<<g128, blk, 0, stream>>>(attbuf + 2048 * 256, Wot, bo, sem_p, alpha_sem, beta_sem,
                                         ln2g, ln2b, combined, nullptr, 256, 256, 768, 0, 256, 1, 0, 1, 0);

    // 7) rsa transform: leaky(LN(rsa_p @ Wrt + brt)) -> combined[:, 512:768]
    gemm_ep<2><<<g128, blk, 0, stream>>>(rsa_p, Wrtt, brt, nullptr, nullptr, nullptr, rtg, rtb,
                                         combined, nullptr, 256, 256, 768, 0, 512, 1, 1, 0, 0);

    // 8) FFN
    gemm_ep<4><<<g128, blk, 0, stream>>>(combined, Wf1t, bf1, nullptr, nullptr, nullptr, f1g, f1b,
                                         h1, nullptr, 768, 768, 512, 0, 0, 1, 1, 0, 0);
    gemm_ep<2><<<g128, blk, 0, stream>>>(h1, Wf2t, bf2, nullptr, nullptr, nullptr, f2g, f2b,
                                         (float*)d_out, nullptr, 512, 512, 256, 0, 0, 1, 1, 0, 0);
}
